// RBM_19920058319444
// MI455X (gfx1250) — matrix operand order repaired and checked by reading, not compiled, not run on hardware
//
#include <hip/hip_runtime.h>
#include <hip/hip_bf16.h>

// ---- CDNA5 WMMA types ----
typedef __attribute__((ext_vector_type(16))) __bf16 v16bf;
typedef __attribute__((ext_vector_type(8)))  __bf16 v8bf;
typedef __attribute__((ext_vector_type(8)))  float  v8f;

union F16 {
    v16bf f;
    struct { v8bf lo, hi; } p;
};

struct BFrag {          // one 32(K)x16(N) tile, hi + lo residual halves
    F16 hi, lo;
};

#define RBM_B 2048
#define RBM_V 4096
#define RBM_H 1024

// ---------------------------------------------------------------------------
// Pack W (fp32, row-major [V][H]) into per-lane WMMA B-fragment layout with
// bf16 hi and bf16 residual lo INTERLEAVED: each lane owns 64 contiguous
// bytes per tile (16 bf16 hi, then 16 bf16 lo).
// Tile (kt,nt) is 32(K)x16(N); lane<16: col=lane, K=kt*32+0..15;
// lane>=16: col=lane-16, K=kt*32+16..31.
// orient=0: B(k,n) = W[k][n]   (v@W,   K over V, N over H)
// orient=1: B(k,n) = W[n][k]   (h@W^T, K over H, N over V; contiguous reads)
// ---------------------------------------------------------------------------
__global__ void pack_w_kernel(const float* __restrict__ W,
                              __bf16* __restrict__ Bp,
                              int KT, int NT, int ldW, int orient) {
    int tid  = blockIdx.x * blockDim.x + threadIdx.x;
    int lane = tid & 31;
    int tile = tid >> 5;
    if (tile >= KT * NT) return;
    int nt = tile % NT;
    int kt = tile / NT;
    int col  = lane & 15;
    int koff = (lane >> 4) * 16;

    F16 fh, fl;
#pragma unroll
    for (int e = 0; e < 16; ++e) {
        int k = kt * 32 + koff + e;
        int n = nt * 16 + col;
        float w = orient ? W[(size_t)n * ldW + k] : W[(size_t)k * ldW + n];
        __bf16 h = (__bf16)w;                 // RNE
        fh.f[e] = h;
        fl.f[e] = (__bf16)(w - (float)h);     // residual
    }
    size_t off = (((size_t)tile) * 32 + lane) * 32;
    *(v8bf*)(Bp + off)      = fh.p.lo;
    *(v8bf*)(Bp + off + 8)  = fh.p.hi;
    *(v8bf*)(Bp + off + 16) = fl.p.lo;
    *(v8bf*)(Bp + off + 24) = fl.p.hi;
}

// ---------------------------------------------------------------------------
// out[row] = visible[row]·b_v   (score init)  + convert visible -> bf16 buffer
// ---------------------------------------------------------------------------
__global__ void rowdot_init_kernel(const float* __restrict__ vis,
                                   const float* __restrict__ bv,
                                   __bf16* __restrict__ vbuf,
                                   float* __restrict__ out, int V) {
    int row = blockIdx.x;
    const float* p = vis + (size_t)row * V;
    __bf16* q = vbuf + (size_t)row * V;
    float acc = 0.f;
    for (int c = threadIdx.x; c < V; c += blockDim.x) {
        float x = p[c];
        q[c] = (__bf16)x;          // exactly 0.0 or 1.0
        acc += x * bv[c];
    }
    __shared__ float red[256];
    red[threadIdx.x] = acc;
    __syncthreads();
    for (int s = 128; s > 0; s >>= 1) {
        if (threadIdx.x < (unsigned)s) red[threadIdx.x] += red[threadIdx.x + s];
        __syncthreads();
    }
    if (threadIdx.x == 0) out[row] = red[0];
}

// ---------------------------------------------------------------------------
__device__ inline BFrag load_bfrag(const __bf16* base) {
    BFrag b;
    b.hi.p.lo = *(const v8bf*)(base);
    b.hi.p.hi = *(const v8bf*)(base + 8);
    b.lo.p.lo = *(const v8bf*)(base + 16);
    b.lo.p.hi = *(const v8bf*)(base + 24);
    return b;
}

__device__ inline F16 load_afrag(const __bf16* pa) {
    F16 a;
    a.p.lo = *(const v8bf*)(pa);
    a.p.hi = *(const v8bf*)(pa + 8);
    return a;
}

// stage fragments for K-step kt_ into buffer set (AA, BB) — all static names
#define STAGE(AA, BB, kt_)                                                    \
    do {                                                                      \
        AA[0] = load_afrag(aRow0 + (size_t)(kt_) * 32);                       \
        AA[1] = load_afrag(aRow1 + (size_t)(kt_) * 32);                       \
        const __bf16* _bk = bBase + (size_t)(kt_) * bKStep;                   \
        BB[0] = load_bfrag(_bk);                                              \
        BB[1] = load_bfrag(_bk + bNStep);                                     \
    } while (0)

// 8 WMMAs on buffer set (AA, BB) — constant indices only
#define MMA_STEP(AA, BB)                                                      \
    do {                                                                      \
        _Pragma("unroll")                                                     \
        for (int mt = 0; mt < 2; ++mt) {                                      \
            _Pragma("unroll")                                                 \
            for (int nt = 0; nt < 2; ++nt) {                                  \
                c[mt][nt] = __builtin_amdgcn_wmma_f32_16x16x32_bf16(          \
                    false, AA[mt].f, false, BB[nt].hi.f, (short)0,            \
                    c[mt][nt], false, false);                                 \
                c[mt][nt] = __builtin_amdgcn_wmma_f32_16x16x32_bf16(          \
                    false, AA[mt].f, false, BB[nt].lo.f, (short)0,            \
                    c[mt][nt], false, false);                                 \
            }                                                                 \
        }                                                                     \
    } while (0)

// ---------------------------------------------------------------------------
// Fused GEMM + bias + sigmoid + Bernoulli + optional score accumulation.
//   pre = bias + A @ (Whi+Wlo) ; s = (sigmoid(pre) > U) ; Out = s (bf16)
//   scoreMode 0: none | 1: Σcols pre*s | 2: Σcols bias*s   (atomicAdd sign*Σ)
// Block = 256 threads (8 wave32), waves tiled 2(M)x4(N).
// Block tile 64(M)x128(N); wave tile 32x32 (2x2 WMMA tiles).
// Manually unrolled ping-pong pipeline (P/Q buffer sets, constant indices
// only, so fragments stay in VGPRs — no dynamic-index scratch lowering).
// ---------------------------------------------------------------------------
__global__ __launch_bounds__(256, 1)
void gemm_rbm_kernel(const __bf16* __restrict__ A,
                     const __bf16* __restrict__ Bpk,
                     const float* __restrict__ bias,
                     const float* __restrict__ U,
                     __bf16* __restrict__ Out,
                     float* __restrict__ score,
                     int N, int Kdim, int scoreMode, float scoreSign) {
    const int lane  = threadIdx.x & 31;
    const int wave  = threadIdx.x >> 5;
    const int mw    = wave & 1;             // 2 wave-rows   -> BM = 64
    const int nw    = wave >> 1;            // 4 wave-cols   -> BN = 128
    const int rowBase = blockIdx.x * 64 + mw * 32;   // wave rows [rowBase, rowBase+32)
    const int ntBase  = blockIdx.y * 8 + nw * 2;     // two global 16-col tiles
    const int NT = N >> 4;
    const int KT = Kdim >> 5;               // even for both shapes (128, 32)
    const int r16  = lane & 15;
    const int hi16 = lane >> 4;

    const __bf16* aRow0 = A + (size_t)(rowBase + r16) * Kdim + hi16 * 16;
    const __bf16* aRow1 = aRow0 + (size_t)16 * Kdim;
    const __bf16* bBase = Bpk + (((size_t)ntBase) * 32 + lane) * 32;
    const size_t bKStep = (size_t)NT * 1024;   // elements per kt step
    const size_t bNStep = 1024;                // elements per 16-col tile

    v8f c[2][2] = {};

    F16   aP[2], aQ[2];
    BFrag bP[2], bQ[2];

    STAGE(aP, bP, 0);

    int kt = 0;
    for (; kt + 2 < KT; kt += 2) {
        __builtin_prefetch((const void*)(bBase + (size_t)(kt + 4) * bKStep), 0, 1);
        STAGE(aQ, bQ, kt + 1);
        MMA_STEP(aP, bP);
        STAGE(aP, bP, kt + 2);
        MMA_STEP(aQ, bQ);
    }
    // tail: kt == KT-2 when KT even
    if (kt + 1 < KT) {
        STAGE(aQ, bQ, kt + 1);
        MMA_STEP(aP, bP);
        MMA_STEP(aQ, bQ);
    } else {
        MMA_STEP(aP, bP);
    }

    // Epilogue. C/D layout: VGPR i -> M = i (lanes 0-15) / i+8 (lanes 16-31); N = lane&15.
#pragma unroll
    for (int mt = 0; mt < 2; ++mt) {
#pragma unroll
        for (int i = 0; i < 8; ++i) {
            int row = rowBase + mt * 16 + hi16 * 8 + i;
            float sacc = 0.f;
#pragma unroll
            for (int nt = 0; nt < 2; ++nt) {
                int col = (ntBase + nt) * 16 + r16;
                float pre = c[mt][nt][i] + bias[col];
                float p   = 1.f / (1.f + __expf(-pre));
                float u   = U[(size_t)row * N + col];
                float s   = (p > u) ? 1.f : 0.f;
                Out[(size_t)row * N + col] = (__bf16)s;
                if (scoreMode == 1)      sacc += pre * s;
                else if (scoreMode == 2) sacc += bias[col] * s;
            }
            if (scoreMode) {
                // reduce the 16 lanes sharing this row (stays inside each half-wave)
                for (int m = 8; m >= 1; m >>= 1) sacc += __shfl_xor(sacc, m, 32);
                if (r16 == 0) atomicAdd(&score[row], scoreSign * sacc);
            }
        }
    }
}

// ---------------------------------------------------------------------------
extern "C" void kernel_launch(void* const* d_in, const int* in_sizes, int n_in,
                              void* d_out, int out_size, void* d_ws, size_t ws_size,
                              hipStream_t stream) {
    const float* visible = (const float*)d_in[0];   // [B,V]
    const float* b_v     = (const float*)d_in[1];   // [V]
    const float* b_h     = (const float*)d_in[2];   // [H]
    const float* W       = (const float*)d_in[3];   // [V,H]
    const float* u_h0    = (const float*)d_in[4];   // [B,H]
    const float* u_v     = (const float*)d_in[5];   // [3,B,V]
    const float* u_h     = (const float*)d_in[6];   // [2,B,H]
    float* out = (float*)d_out;                     // [B]

    const int B = RBM_B, V = RBM_V, H = RBM_H;
    (void)in_sizes; (void)n_in; (void)out_size; (void)ws_size;

    char* ws = (char*)d_ws;
    __bf16* WA_pk = (__bf16*)ws; ws += (size_t)V * H * 4;   // v@W  packed hi+lo
    __bf16* WB_pk = (__bf16*)ws; ws += (size_t)V * H * 4;   // h@W^T packed hi+lo
    __bf16* vbuf  = (__bf16*)ws; ws += (size_t)B * V * 2;   // v activations (bf16)
    __bf16* hbuf  = (__bf16*)ws; ws += (size_t)B * H * 2;   // h activations (bf16)

    // Pack W both orientations (hi/lo split, interleaved).
    {
        int ktA = V / 32, ntA = H / 16;   // v@W : K over V, N over H
        int ktB = H / 32, ntB = V / 16;   // h@W^T: K over H, N over V
        pack_w_kernel<<<(ktA * ntA * 32) / 256, 256, 0, stream>>>(W, WA_pk, ktA, ntA, H, 0);
        pack_w_kernel<<<(ktB * ntB * 32) / 256, 256, 0, stream>>>(W, WB_pk, ktB, ntB, H, 1);
    }

    // score init: out[b] = visible·b_v ; also visible -> bf16
    rowdot_init_kernel<<<B, 256, 0, stream>>>(visible, b_v, vbuf, out, V);

    dim3 blk(256);
    dim3 gA(B / 64, H / 128);   // outputs [B,H]
    dim3 gB(B / 64, V / 128);   // outputs [B,V]
    const size_t sv = (size_t)B * V, sh = (size_t)B * H;

    // positive phase: h0 = bern(sig(b_h + vis@W), u_h0); out += Σ pre*h
    gemm_rbm_kernel<<<gA, blk, 0, stream>>>(vbuf, WA_pk, b_h, u_h0, hbuf, out, H, V, 1, +1.f);
    // k=0: v1 = bern(sig(b_v + h0@W^T), u_v[0])
    gemm_rbm_kernel<<<gB, blk, 0, stream>>>(hbuf, WB_pk, b_v, u_v + 0 * sv, vbuf, out, V, H, 0, 0.f);
    // k=1: h2 = bern(sig(b_h + v1@W), u_h[0])
    gemm_rbm_kernel<<<gA, blk, 0, stream>>>(vbuf, WA_pk, b_h, u_h + 0 * sh, hbuf, out, H, V, 0, 0.f);
    // k=2: v3 = bern(sig(b_v + h2@W^T), u_v[1])
    gemm_rbm_kernel<<<gB, blk, 0, stream>>>(hbuf, WB_pk, b_v, u_v + 1 * sv, vbuf, out, V, H, 0, 0.f);
    // k=3: h3 = bern(sig(b_h + v3@W), u_h[1]); out -= h3·b_h
    gemm_rbm_kernel<<<gA, blk, 0, stream>>>(vbuf, WA_pk, b_h, u_h + 1 * sh, hbuf, out, H, V, 2, -1.f);
    // k=4: v5 = bern(sig(b_v + h3@W^T), u_v[2]); out -= Σ v5*pre
    gemm_rbm_kernel<<<gB, blk, 0, stream>>>(hbuf, WB_pk, b_v, u_v + 2 * sv, vbuf, out, V, H, 1, -1.f);
}
